// StackedGATv2_68891275427971
// MI455X (gfx1250) — matrix operand order repaired and checked by reading, not compiled, not run on hardware
//
#include <hip/hip_runtime.h>
#include <hip/hip_bf16.h>
#include <math.h>

typedef __attribute__((ext_vector_type(16))) _Float16 v16h;
typedef __attribute__((ext_vector_type(8)))  float    v8f;
typedef __attribute__((ext_vector_type(4)))  unsigned int v4u;
typedef __attribute__((ext_vector_type(8)))  int      v8i_;
typedef __attribute__((ext_vector_type(4)))  int      v4i_;

#define NEG_SLOPE 0.2f
#define LN_EPS 1e-5f

// ===========================================================================
// WMMA fragment layouts (wave32, 16x16x32 f16), per CDNA5 ISA 7.12.2:
//  A (16Mx32K): lane L -> row M=L&15; lanes 0-15: halves 0..7 = K0..7,
//    halves 8..15 = K16..23; lanes 16-31: K8..15 then K24..31.
//  B (32Kx16N): lane L -> col n=L&15; same K order as A: lanes 0-15:
//    halves 0..7 = K0..7, 8..15 = K16..23; lanes 16-31: K8..15, K24..31.
// Fragments are PRE-PACKED so each lane's v16h is a contiguous 32B chunk:
// LDS reads/writes are 2x ds_load/store_b128, global B reads 2x b128.
// ===========================================================================

// ---------------------------------------------------------------------------
// Pack a KxD (row-major f32) weight matrix into B-fragment order (f16).
// ---------------------------------------------------------------------------
__global__ void pack_b_kernel(const float* __restrict__ W,
                              _Float16* __restrict__ out, int K, int D) {
  int j = blockIdx.x * blockDim.x + threadIdx.x;
  int total = (K >> 5) * (D >> 4) * 32;
  if (j >= total) return;
  int lane = j & 31, tile = j >> 5;
  int ntn = D >> 4;
  int ks = tile / ntn, nt = tile - ks * ntn;
  int n  = nt * 16 + (lane & 15);
  int k0 = ks * 32 + ((lane >> 4) << 3);
  v16h b;
#pragma unroll
  for (int e = 0; e < 16; ++e) b[e] = (_Float16)W[(size_t)(k0 + e + (e & 8)) * D + n];
  ((v16h*)out)[j] = b;
}

// ---------------------------------------------------------------------------
// Encoder: h = relu(x @ W_enc + b_enc). x:[N,128], W packed, D=64.
// 256 thr = 8 waves; block tile 32(M)x64(N); wave tile 16x16; K=128 (4 steps).
// ---------------------------------------------------------------------------
__global__ __launch_bounds__(256) void encoder_kernel(
    const float* __restrict__ x, const _Float16* __restrict__ Bp,
    const float* __restrict__ bias, float* __restrict__ h, int N) {
  constexpr int F = 128, D = 64, KS = 4;
  __shared__ v16h sAv[2 * KS * 32];           // 8 tiles x 32 lanes x 32B = 8 KB
  const int tid = threadIdx.x, lane = tid & 31, wave = tid >> 5;
  const int rowBase = blockIdx.x * 32;
  const bool full = (rowBase + 32) <= N;      // block-uniform: no guards needed

  { // stage A fragments: exactly 256 fragment-builder threads
    int tile = tid >> 5, ln = tid & 31;
    int smt = tile >> 2, sks = tile & 3;
    int m = ln & 15, kh = (ln >> 4) << 3;
    int row = rowBase + smt * 16 + m;
    v16h a;
    if (full) {
      const float* xs = x + (size_t)row * F + sks * 32;
#pragma unroll
      for (int e = 0; e < 8; ++e) {
        a[e]     = (_Float16)xs[kh + e];
        a[e + 8] = (_Float16)xs[16 + kh + e];
      }
    } else {
      bool ok = row < N;
      const float* xs = x + (size_t)(ok ? row : 0) * F + sks * 32;
#pragma unroll
      for (int e = 0; e < 8; ++e) {
        a[e]     = (_Float16)(ok ? xs[kh + e] : 0.0f);
        a[e + 8] = (_Float16)(ok ? xs[16 + kh + e] : 0.0f);
      }
    }
    sAv[tile * 32 + ln] = a;                  // 2x ds_store_b128
  }
  __syncthreads();

  const int mt = wave >> 2, nt = wave & 3;
  const v16h* Bv = (const v16h*)Bp;
  v8f acc = {};
#pragma unroll
  for (int ks = 0; ks < KS; ++ks) {
    v16h a = sAv[(mt * KS + ks) * 32 + lane];   // 2x ds_load_b128
    v16h b = Bv[(ks * 4 + nt) * 32 + lane];     // 2x global_load_b128 (L2-hot)
    acc = __builtin_amdgcn_wmma_f32_16x16x32_f16(false, a, false, b,
                                                 (short)0, acc, false, false);
  }
  const int col = nt * 16 + (lane & 15);
  const float bv = bias[col];
  float* hp = h + (size_t)(rowBase + mt * 16 + ((lane >> 4) << 3)) * D + col;
  if (full) {
#pragma unroll
    for (int r = 0; r < 8; ++r) {
      float v = acc[r] + bv;
      hp[(size_t)r * D] = v > 0.0f ? v : 0.0f;
    }
  } else {
#pragma unroll
    for (int r = 0; r < 8; ++r) {
      int row = rowBase + mt * 16 + r + ((lane >> 4) << 3);
      if (row < N) {
        float v = acc[r] + bv;
        h[(size_t)row * D + col] = v > 0.0f ? v : 0.0f;
      }
    }
  }
}

// ---------------------------------------------------------------------------
// Layer linear: xl = h@Wl+bl, xr = h@Wr+br (D=64, K=64, 2 k-steps).
// h tile staged via Tensor Data Mover (tensor_load_to_lds; HW OOB zero-fill
// handles the tail block), converted in-LDS to packed f16 fragments;
// weights pre-packed in global (L2-resident).
// ---------------------------------------------------------------------------
__global__ __launch_bounds__(256) void layer_lin_kernel(
    const float* __restrict__ h,
    const _Float16* __restrict__ BpL, const float* __restrict__ bl,
    const _Float16* __restrict__ BpR, const float* __restrict__ br,
    float* __restrict__ xl, float* __restrict__ xr, int N) {
  constexpr int D = 64, KS = 2;
  __shared__ float sRaw[32 * D];              // 8 KB raw f32 tile (TDM target)
  __shared__ v16h  sAv[2 * KS * 32];          // 4 KB packed fragments
  const int tid = threadIdx.x, lane = tid & 31, wave = tid >> 5;
  const int rowBase = blockIdx.x * 32;
  const bool full = (rowBase + 32) <= N;

#if __has_builtin(__builtin_amdgcn_tensor_load_to_lds) && \
    __has_builtin(__builtin_amdgcn_s_wait_tensorcnt)
  if (tid < 32) {   // one wave issues the TDM op (EXEC ignored; TENSORcnt per-wave)
    unsigned long long ga = (unsigned long long)(uintptr_t)(h + (size_t)rowBase * D);
    unsigned int galo = (unsigned int)ga;
    unsigned int gahi = (unsigned int)((ga >> 32) & 0x1FFFFFFull);
    unsigned int ldsa = (unsigned int)(uintptr_t)(void*)sRaw; // low 32b = LDS offset
    unsigned int rows = (unsigned int)(N - rowBase);          // HW OOB -> zero fill
    // D# group0: count=1 | lds_addr | global_addr | type=2 (bits 127:126)
    v4u g0 = { 1u, ldsa, galo, gahi | (2u << 30) };
    // D# group1: data_size=4B, tensor_dim0=D, tensor_dim1=rows,
    //            tile_dim0=D, tile_dim1=32, tensor_dim0_stride=D
    v8i_ g1;
    g1[0] = (int)(2u << 16);                                  // data_size=2 (4B)
    g1[1] = (int)(((unsigned)D & 0xFFFFu) << 16);             // dim0[15:0] @ [63:48]
    g1[2] = (int)((((unsigned)D) >> 16) | ((rows & 0xFFFFu) << 16));
    g1[3] = (int)((rows >> 16) | ((unsigned)D << 16));        // dim1 hi | tile_dim0
    g1[4] = 32;                                               // tile_dim1=32
    g1[5] = D;                                                // dim0_stride lo32
    g1[6] = 0; g1[7] = 0;
    v4i_ gz4 = {0, 0, 0, 0};
    v8i_ gz8 = {0, 0, 0, 0, 0, 0, 0, 0};
    __builtin_amdgcn_tensor_load_to_lds(g0, g1, gz4, gz4, gz8, 0);
    __builtin_amdgcn_s_wait_tensorcnt(0);
  }
  __syncthreads();
  if (tid < 2 * KS * 32) {                   // 128 fragment builders from LDS
    int tile = tid >> 5, ln = tid & 31;
    int smt = tile >> 1, sks = tile & 1;
    int m = ln & 15, kh = (ln >> 4) << 3;
    const float* src = &sRaw[(smt * 16 + m) * D + sks * 32];
    v16h a;
#pragma unroll
    for (int e = 0; e < 8; ++e) {
      a[e]     = (_Float16)src[kh + e];
      a[e + 8] = (_Float16)src[16 + kh + e];
    }
    sAv[tile * 32 + ln] = a;
  }
#else
  if (tid < 2 * KS * 32) {                   // fallback: gather from global
    int tile = tid >> 5, ln = tid & 31;
    int smt = tile >> 1, sks = tile & 1;
    int m = ln & 15, kh = (ln >> 4) << 3;
    int row = rowBase + smt * 16 + m;
    bool ok = row < N;
    const float* src = h + (size_t)(ok ? row : 0) * D + sks * 32;
    v16h a;
#pragma unroll
    for (int e = 0; e < 8; ++e) {
      a[e]     = (_Float16)(ok ? src[kh + e] : 0.0f);
      a[e + 8] = (_Float16)(ok ? src[16 + kh + e] : 0.0f);
    }
    sAv[tile * 32 + ln] = a;
  }
#endif
  __syncthreads();

  const int mt = wave >> 2, nt = wave & 3;
  const v16h* BvL = (const v16h*)BpL;
  const v16h* BvR = (const v16h*)BpR;
  v8f accL = {}, accR = {};
#pragma unroll
  for (int ks = 0; ks < KS; ++ks) {
    v16h a  = sAv[(mt * KS + ks) * 32 + lane];
    v16h bL = BvL[(ks * 4 + nt) * 32 + lane];
    v16h bR = BvR[(ks * 4 + nt) * 32 + lane];
    accL = __builtin_amdgcn_wmma_f32_16x16x32_f16(false, a, false, bL,
                                                  (short)0, accL, false, false);
    accR = __builtin_amdgcn_wmma_f32_16x16x32_f16(false, a, false, bR,
                                                  (short)0, accR, false, false);
  }
  const int col = nt * 16 + (lane & 15);
  const float bLv = bl[col], bRv = br[col];
  const size_t base = (size_t)(rowBase + mt * 16 + ((lane >> 4) << 3)) * D + col;
  if (full) {
#pragma unroll
    for (int r = 0; r < 8; ++r) {
      xl[base + (size_t)r * D] = accL[r] + bLv;
      xr[base + (size_t)r * D] = accR[r] + bRv;
    }
  } else {
#pragma unroll
    for (int r = 0; r < 8; ++r) {
      int row = rowBase + mt * 16 + r + ((lane >> 4) << 3);
      if (row < N) {
        xl[(size_t)row * D + col] = accL[r] + bLv;
        xr[(size_t)row * D + col] = accR[r] + bRv;
      }
    }
  }
}

// ---------------------------------------------------------------------------
// Per-layer init: smax=-inf, denom=0, agg=0
// ---------------------------------------------------------------------------
__global__ void init_layer_kernel(float* smax, float* denom, float* agg, int N) {
  int i = blockIdx.x * blockDim.x + threadIdx.x;
  if (i < N * 4) { smax[i] = -INFINITY; denom[i] = 0.0f; }
  if (i < N * 64) agg[i] = 0.0f;
}

__device__ __forceinline__ void atomicMaxF32(float* addr, float v) {
  if (v >= 0.0f) atomicMax((int*)addr, __float_as_int(v));
  else           atomicMin((unsigned int*)addr, __float_as_uint(v));
}

// ---------------------------------------------------------------------------
// Edge pass 1: score = att . leaky_relu(xl[src]+xr[dst]); segment max (atomic)
// ---------------------------------------------------------------------------
__global__ __launch_bounds__(256) void edge_score_kernel(
    const int* __restrict__ ei, int E, int N,
    const float* __restrict__ xl, const float* __restrict__ xr,
    const float* __restrict__ att, float* __restrict__ score,
    float* __restrict__ smax) {
  const int Et = E + N;
  int t = blockIdx.x * blockDim.x + threadIdx.x;
  if (t >= Et * 4) return;
  int e = t >> 2, hh = t & 3;
  __builtin_prefetch(&ei[e + 8192], 0, 1);   // global_prefetch_b8 on index stream
  int s, dv;
  if (e < E) { s = ei[e]; dv = ei[E + e]; } else { s = dv = e - E; }
  const float4* xl4  = (const float4*)xl;
  const float4* xr4  = (const float4*)xr;
  const float4* att4 = (const float4*)att;
  float sc = 0.0f;
#pragma unroll
  for (int q = 0; q < 4; ++q) {
    float4 a = xl4[(size_t)s * 16 + hh * 4 + q];
    float4 b = xr4[(size_t)dv * 16 + hh * 4 + q];
    float4 w = att4[hh * 4 + q];
    float v;
    v = a.x + b.x; v = v > 0.0f ? v : NEG_SLOPE * v; sc = fmaf(v, w.x, sc);
    v = a.y + b.y; v = v > 0.0f ? v : NEG_SLOPE * v; sc = fmaf(v, w.y, sc);
    v = a.z + b.z; v = v > 0.0f ? v : NEG_SLOPE * v; sc = fmaf(v, w.z, sc);
    v = a.w + b.w; v = v > 0.0f ? v : NEG_SLOPE * v; sc = fmaf(v, w.w, sc);
  }
  score[t] = sc;
  atomicMaxF32(&smax[dv * 4 + hh], sc);
}

// ---------------------------------------------------------------------------
// Edge pass 2: ex = exp(score - smax[dst]); segment sum (atomic)
// ---------------------------------------------------------------------------
__global__ __launch_bounds__(256) void edge_exp_kernel(
    const int* __restrict__ ei, int E, int N,
    const float* __restrict__ smax, float* __restrict__ score,
    float* __restrict__ denom) {
  const int Et = E + N;
  int t = blockIdx.x * blockDim.x + threadIdx.x;
  if (t >= Et * 4) return;
  int e = t >> 2, hh = t & 3;
  int dv = (e < E) ? ei[E + e] : (e - E);
  float ex = __expf(score[t] - smax[dv * 4 + hh]);
  score[t] = ex;
  atomicAdd(&denom[dv * 4 + hh], ex);
}

// ---------------------------------------------------------------------------
// Edge pass 3: agg[dst] += (ex/denom[dst]) * xl[src]
// ---------------------------------------------------------------------------
__global__ __launch_bounds__(256) void edge_agg_kernel(
    const int* __restrict__ ei, int E, int N,
    const float* __restrict__ xl, const float* __restrict__ score,
    const float* __restrict__ denom, float* __restrict__ agg) {
  const int Et = E + N;
  int t = blockIdx.x * blockDim.x + threadIdx.x;
  if (t >= Et * 4) return;
  int e = t >> 2, hh = t & 3;
  int s, dv;
  if (e < E) { s = ei[e]; dv = ei[E + e]; } else { s = dv = e - E; }
  float alpha = score[t] / denom[dv * 4 + hh];
  const float* xs = &xl[(size_t)s * 64 + hh * 16];
  float* ad = &agg[(size_t)dv * 64 + hh * 16];
#pragma unroll
  for (int i = 0; i < 16; ++i) atomicAdd(&ad[i], alpha * xs[i]);
}

// ---------------------------------------------------------------------------
// LayerNorm + ELU: one wave32 per node (64 values = 2 per lane).
// ---------------------------------------------------------------------------
__global__ __launch_bounds__(256) void ln_elu_kernel(
    const float* __restrict__ agg, const float* __restrict__ out_bias,
    const float* __restrict__ gamma, const float* __restrict__ beta,
    float* __restrict__ out, int N) {
  int node = (blockIdx.x * blockDim.x + threadIdx.x) >> 5;
  int lane = threadIdx.x & 31;
  if (node >= N) return;
  float v0 = agg[(size_t)node * 64 + lane]      + out_bias[lane];
  float v1 = agg[(size_t)node * 64 + 32 + lane] + out_bias[32 + lane];
  float s = v0 + v1;
#pragma unroll
  for (int off = 16; off > 0; off >>= 1) s += __shfl_xor(s, off, 32);
  float mu = s * (1.0f / 64.0f);
  float d0 = v0 - mu, d1 = v1 - mu;
  float vs = d0 * d0 + d1 * d1;
#pragma unroll
  for (int off = 16; off > 0; off >>= 1) vs += __shfl_xor(vs, off, 32);
  float rstd = rsqrtf(vs * (1.0f / 64.0f) + LN_EPS);
  float y0 = d0 * rstd * gamma[lane]      + beta[lane];
  float y1 = d1 * rstd * gamma[32 + lane] + beta[32 + lane];
  out[(size_t)node * 64 + lane]      = y0 > 0.0f ? y0 : expm1f(y0);
  out[(size_t)node * 64 + 32 + lane] = y1 > 0.0f ? y1 : expm1f(y1);
}

// ---------------------------------------------------------------------------
// Host-side orchestration
// ---------------------------------------------------------------------------
extern "C" void kernel_launch(void* const* d_in, const int* in_sizes, int n_in,
                              void* d_out, int out_size, void* d_ws, size_t ws_size,
                              hipStream_t stream) {
  const float* x     = (const float*)d_in[0];
  const int*   ei    = (const int*)d_in[1];
  const float* W_enc = (const float*)d_in[2];
  const float* b_enc = (const float*)d_in[3];
  const float* Wl    = (const float*)d_in[4];
  const float* bl    = (const float*)d_in[5];
  const float* Wr    = (const float*)d_in[6];
  const float* br    = (const float*)d_in[7];
  const float* att   = (const float*)d_in[8];
  const float* ob    = (const float*)d_in[9];
  const float* gamma = (const float*)d_in[10];
  const float* beta  = (const float*)d_in[11];

  const int D = in_sizes[3];               // 64
  const int F = in_sizes[2] / D;           // 128
  const int N = in_sizes[0] / F;           // 100000
  const int E = in_sizes[1] / 2;           // 1600000
  const int L = in_sizes[5] / D;           // 3
  const int Et = E + N;

  float* ws    = (float*)d_ws;
  float* h     = ws;                        // N*D
  float* xl    = h    + (size_t)N * D;
  float* xr    = xl   + (size_t)N * D;
  float* agg   = xr   + (size_t)N * D;
  float* score = agg  + (size_t)N * D;      // Et*4
  float* smax  = score + (size_t)Et * 4;    // N*4
  float* denom = smax + (size_t)N * 4;      // N*4
  _Float16* encWp = (_Float16*)(denom + (size_t)N * 4);  // F*D halves (16B aligned)
  _Float16* Wp    = encWp + (size_t)F * D;               // L*2*D*D halves

  const int gemmBlocks = (N + 31) / 32;
  const int edgeBlocks = (Et * 4 + 255) / 256;
  const int initBlocks = (N * 64 + 255) / 256;
  const int lnBlocks   = (N * 32 + 255) / 256;

  // Pack weights into WMMA B-fragment order (runs once per call, L2-resident)
  pack_b_kernel<<<(F * D / 16 + 255) / 256, 256, 0, stream>>>(W_enc, encWp, F, D);
  for (int l = 0; l < L; ++l) {
    pack_b_kernel<<<(D * D / 16 + 255) / 256, 256, 0, stream>>>(
        Wl + (size_t)l * D * D, Wp + (size_t)(2 * l) * D * D, D, D);
    pack_b_kernel<<<(D * D / 16 + 255) / 256, 256, 0, stream>>>(
        Wr + (size_t)l * D * D, Wp + (size_t)(2 * l + 1) * D * D, D, D);
  }

  encoder_kernel<<<gemmBlocks, 256, 0, stream>>>(x, encWp, b_enc, h, N);

  for (int l = 0; l < L; ++l) {
    const float* bl_l  = bl  + (size_t)l * D;
    const float* br_l  = br  + (size_t)l * D;
    const float* att_l = att + (size_t)l * D;   // H*dh == D
    const float* ob_l  = ob  + (size_t)l * D;
    const float* ga_l  = gamma + (size_t)l * D;
    const float* be_l  = beta  + (size_t)l * D;
    float* out_l = (l == L - 1) ? (float*)d_out : h;

    layer_lin_kernel<<<gemmBlocks, 256, 0, stream>>>(
        h, Wp + (size_t)(2 * l) * D * D, bl_l,
        Wp + (size_t)(2 * l + 1) * D * D, br_l, xl, xr, N);
    init_layer_kernel<<<initBlocks, 256, 0, stream>>>(smax, denom, agg, N);
    edge_score_kernel<<<edgeBlocks, 256, 0, stream>>>(ei, E, N, xl, xr, att_l,
                                                      score, smax);
    edge_exp_kernel<<<edgeBlocks, 256, 0, stream>>>(ei, E, N, smax, score, denom);
    edge_agg_kernel<<<edgeBlocks, 256, 0, stream>>>(ei, E, N, xl, score, denom, agg);
    ln_elu_kernel<<<lnBlocks, 256, 0, stream>>>(agg, ob_l, ga_l, be_l, out_l, N);
  }
}
